// Attention_20358144983545
// MI455X (gfx1250) — matrix operand order repaired and checked by reading, not compiled, not run on hardware
//
#include <hip/hip_runtime.h>
#include <hip/hip_bf16.h>
#include <cstdint>

// ---------------------------------------------------------------------------
// Types for CDNA5 WMMA (wave32, 16x16x32 bf16 -> f32)
// ---------------------------------------------------------------------------
typedef __bf16 bf16;
typedef __attribute__((ext_vector_type(16))) __bf16 v16bf;
typedef __attribute__((ext_vector_type(8)))  __bf16 v8bf;
typedef __attribute__((ext_vector_type(8)))  float  v8f;

#define DIMX      512
#define HEADS     8
#define DHEAD     64
#define SEQ       1024
#define BATCH     8
#define ROWS      (BATCH * SEQ)        // 8192 tokens
#define NQKV      (3 * HEADS * DHEAD)  // 1536

// Compiler-only memory fence: CDNA5 LDS ops from one wave execute in order
// (ISA 7.3 "LDS: in order"), so wave-private LDS staging needs no s_barrier,
// only protection against compiler reordering.
#define WAVE_MEM_FENCE() asm volatile("" ::: "memory")

__device__ __forceinline__ v8f wmma_bf16(v16bf a, v16bf b, v8f c) {
    return __builtin_amdgcn_wmma_f32_16x16x32_bf16(
        /*neg_a=*/false, a, /*neg_b=*/false, b,
        /*c_mod=*/(short)0, c, /*reuse_a=*/false, /*reuse_b=*/false);
}

// A fragment: 16x32 bf16, row-major source, ld elements per row.
// lanes 0-15: row=l, K = k0..k0+7  and k0+16..k0+23
// lanes 16-31: row=l, K = k0+8..k0+15 and k0+24..k0+31
__device__ __forceinline__ v16bf load_a_frag(const bf16* base, int ld, int k0, int lane) {
    int l = lane & 15, hi = lane >> 4;
    const bf16* p = base + (size_t)l * ld + k0 + hi * 8;
    v8bf lo = *(const v8bf*)(p);
    v8bf hh = *(const v8bf*)(p + 16);
    return __builtin_shufflevector(lo, hh, 0,1,2,3,4,5,6,7,8,9,10,11,12,13,14,15);
}

// B fragment: 32x16 bf16 from a transposed ("N-major, K-contiguous") source.
// lane: column n0+(lane&15), K = k0+8*hi..+7 and k0+16+8*hi..+7 (hi = lane>=16).
__device__ __forceinline__ v16bf load_b_frag(const bf16* baseT, int ld, int n0, int k0, int lane) {
    int l = lane & 15, hi = lane >> 4;
    const bf16* p = baseT + (size_t)(n0 + l) * ld + k0 + hi * 8;
    v8bf lo = *(const v8bf*)(p);
    v8bf hh = *(const v8bf*)(p + 16);
    return __builtin_shufflevector(lo, hh, 0,1,2,3,4,5,6,7,8,9,10,11,12,13,14,15);
}

// ---------------------------------------------------------------------------
// Kernel 1: L2-normalize rows of x (scale sqrt(512)*gamma), write bf16 xn,
//           and compute the 8-wide gate GEMV + sigmoid inline.
// 1 block (128 thr) per token row.
// ---------------------------------------------------------------------------
__global__ __launch_bounds__(128)
void norm_gates_kernel(const float* __restrict__ x, const float* __restrict__ gamma,
                       const float* __restrict__ wg, const float* __restrict__ bg,
                       bf16* __restrict__ xn_bf, float* __restrict__ gates) {
    int row  = blockIdx.x;
    int t    = threadIdx.x;
    int wave = t >> 5, lane = t & 31;
    const float* xr = x + (size_t)row * DIMX;

    float4 xv = ((const float4*)xr)[t];
    float ss = xv.x*xv.x + xv.y*xv.y + xv.z*xv.z + xv.w*xv.w;
    #pragma unroll
    for (int off = 1; off < 32; off <<= 1) ss += __shfl_xor(ss, off, 32);

    __shared__ float red[4];
    __shared__ float gred[4][HEADS];
    if (lane == 0) red[wave] = ss;
    __syncthreads();
    float tot  = red[0] + red[1] + red[2] + red[3];
    float nrm  = fmaxf(sqrtf(tot), 1e-12f);
    float scal = 22.62741699796952f / nrm;   // sqrt(512)/||x||

    int c0 = t * 4;
    float xn0 = xv.x * scal * gamma[c0 + 0];
    float xn1 = xv.y * scal * gamma[c0 + 1];
    float xn2 = xv.z * scal * gamma[c0 + 2];
    float xn3 = xv.w * scal * gamma[c0 + 3];
    bf16* xd = xn_bf + (size_t)row * DIMX + c0;
    xd[0] = (bf16)xn0; xd[1] = (bf16)xn1; xd[2] = (bf16)xn2; xd[3] = (bf16)xn3;

    float pg[HEADS];
    #pragma unroll
    for (int h = 0; h < HEADS; ++h) {
        pg[h] = xn0 * wg[(c0 + 0) * HEADS + h]
              + xn1 * wg[(c0 + 1) * HEADS + h]
              + xn2 * wg[(c0 + 2) * HEADS + h]
              + xn3 * wg[(c0 + 3) * HEADS + h];
        #pragma unroll
        for (int off = 1; off < 32; off <<= 1) pg[h] += __shfl_xor(pg[h], off, 32);
    }
    if (lane == 0) {
        #pragma unroll
        for (int h = 0; h < HEADS; ++h) gred[wave][h] = pg[h];
    }
    __syncthreads();
    if (t < HEADS) {
        float s = gred[0][t] + gred[1][t] + gred[2][t] + gred[3][t] + bg[t];
        gates[(size_t)row * HEADS + t] = 1.0f / (1.0f + __expf(-s));
    }
}

// ---------------------------------------------------------------------------
// Kernel 2/3: transpose + f32->bf16 weights:  w[R][C] -> wT[C][R]
// ---------------------------------------------------------------------------
__global__ __launch_bounds__(256)
void transpose_w_kernel(const float* __restrict__ w, bf16* __restrict__ wT, int R, int C) {
    int idx = blockIdx.x * 256 + threadIdx.x;
    if (idx >= R * C) return;
    int r = idx / C, c = idx % C;
    wT[(size_t)c * R + r] = (bf16)w[(size_t)r * C + c];
}

// ---------------------------------------------------------------------------
// Kernel 4/7: bf16 WMMA GEMM  C[M][N] = A[M][K] * BT[N][K]^T
// 8 waves per block in a 4(M) x 2(N) arrangement; each wave owns a
// 32(M) x 64(N) tile: 2 A-frags x 4 B-frags -> 8 WMMAs per k-step with the
// B fragment reused across both M-tiles (halves redundant B traffic).
// ---------------------------------------------------------------------------
template <bool F32OUT>
__global__ __launch_bounds__(256)
void gemm_bf16_kernel(const bf16* __restrict__ A, const bf16* __restrict__ BT,
                      void* __restrict__ Cout, int M, int N, int K) {
    int wave = threadIdx.x >> 5;
    int lane = threadIdx.x & 31;
    int wm   = wave & 3, wn = wave >> 2;
    int row0 = blockIdx.x * 128 + wm * 32;
    int col0 = blockIdx.y * 128 + wn * 64;

    v8f acc[2][4] = {};
    const bf16* arow0 = A + (size_t)row0 * K;
    const bf16* arow1 = A + (size_t)(row0 + 16) * K;
    int l = lane & 15, hi = lane >> 4;

    for (int k0 = 0; k0 < K; k0 += 32) {
        if (k0 + 32 < K) {
            // speculative prefetch of the next k-slice (global_prefetch_b8)
            __builtin_prefetch((const void*)(arow0 + (size_t)l * K + k0 + 32), 0, 3);
            __builtin_prefetch((const void*)(BT + (size_t)(col0 + l) * K + k0 + 32), 0, 3);
        }
        v16bf a0 = load_a_frag(arow0, K, k0, lane);
        v16bf a1 = load_a_frag(arow1, K, k0, lane);
        #pragma unroll
        for (int j = 0; j < 4; ++j) {
            v16bf b = load_b_frag(BT, K, col0 + j * 16, k0, lane);
            acc[0][j] = wmma_bf16(a0, b, acc[0][j]);
            acc[1][j] = wmma_bf16(a1, b, acc[1][j]);
        }
    }

    if (F32OUT) {
        float* C = (float*)Cout;
        #pragma unroll
        for (int m = 0; m < 2; ++m)
            #pragma unroll
            for (int j = 0; j < 4; ++j)
                #pragma unroll
                for (int r = 0; r < 8; ++r)
                    C[(size_t)(row0 + m * 16 + r + 8 * hi) * N + col0 + j * 16 + l] = acc[m][j][r];
    } else {
        bf16* C = (bf16*)Cout;
        #pragma unroll
        for (int m = 0; m < 2; ++m)
            #pragma unroll
            for (int j = 0; j < 4; ++j)
                #pragma unroll
                for (int r = 0; r < 8; ++r)
                    C[(size_t)(row0 + m * 16 + r + 8 * hi) * N + col0 + j * 16 + l] = (bf16)acc[m][j][r];
    }
}

// ---------------------------------------------------------------------------
// Kernel 5: RoPE on q,k + split qkv into attention layouts.
//   q,k : [b*h][n][64]   (row-major; A-frag / K-as-B-frag contiguous)
//   vT  : [b*h][64][n]   (d-major; P*V B-frag contiguous in keys)
// One thread per (b,h,n, pair i in 0..31).
// ---------------------------------------------------------------------------
__global__ __launch_bounds__(256)
void rope_split_kernel(const bf16* __restrict__ qkv, bf16* __restrict__ q,
                       bf16* __restrict__ k, bf16* __restrict__ vT) {
    unsigned idx = blockIdx.x * 256 + threadIdx.x;     // 2^21 total
    int i  = idx & 31;
    int nn = (idx >> 5) & 1023;
    int h  = (idx >> 15) & 7;
    int b  = idx >> 18;

    const bf16* row = qkv + ((size_t)b * SEQ + nn) * NQKV;
    int c0 = h * DHEAD + 2 * i;

    float fexp = (float)(2 * i) / (float)DHEAD;
    float invf = __powf(10000.0f, -fexp);
    float ang  = (float)nn * invf;
    float cs = __cosf(ang), sn = __sinf(ang);

    float q0 = (float)row[c0],           q1 = (float)row[c0 + 1];
    float k0 = (float)row[512 + c0],     k1 = (float)row[512 + c0 + 1];
    float v0 = (float)row[1024 + c0],    v1 = (float)row[1024 + c0 + 1];

    size_t bh = (size_t)b * HEADS + h;
    bf16* qd = q + bh * SEQ * DHEAD + (size_t)nn * DHEAD + 2 * i;
    bf16* kd = k + bh * SEQ * DHEAD + (size_t)nn * DHEAD + 2 * i;
    qd[0] = (bf16)(q0 * cs - q1 * sn);
    qd[1] = (bf16)(q1 * cs + q0 * sn);
    kd[0] = (bf16)(k0 * cs - k1 * sn);
    kd[1] = (bf16)(k1 * cs + k0 * sn);

    bf16* vd = vT + bh * DHEAD * SEQ;
    vd[(size_t)(2 * i)     * SEQ + nn] = (bf16)v0;
    vd[(size_t)(2 * i + 1) * SEQ + nn] = (bf16)v1;
}

// ---------------------------------------------------------------------------
// Kernel 6: flash attention per (b,h), 4 waves x 16 queries, 64-key blocks.
// S = QK^T via WMMA (8 per block); online softmax with per-half shfl row
// reductions; P restaged through *wave-private* LDS (no workgroup barrier
// needed -- LDS ops are in-order within a wave) into A-fragments; O += P*V
// via WMMA (8 per block). Epilogue fuses 1/l and the sigmoid gate.
// ---------------------------------------------------------------------------
__global__ __launch_bounds__(128)
void attn_kernel(const bf16* __restrict__ q, const bf16* __restrict__ k,
                 const bf16* __restrict__ vT, const float* __restrict__ gates,
                 bf16* __restrict__ out) {
    int bh   = blockIdx.y;            // b*8 + h
    int b    = bh >> 3, h = bh & 7;
    int wave = threadIdx.x >> 5;
    int lane = threadIdx.x & 31;
    int q0r  = blockIdx.x * 64 + wave * 16;

    const bf16* qb = q  + (size_t)bh * SEQ * DHEAD;
    const bf16* kb = k  + (size_t)bh * SEQ * DHEAD;
    const bf16* vb = vT + (size_t)bh * DHEAD * SEQ;

    __shared__ alignas(32) bf16 plds[4][16 * 64];   // wave-private P staging
    bf16* pl = plds[wave];

    v16bf qa0 = load_a_frag(qb + (size_t)q0r * DHEAD, DHEAD, 0,  lane);
    v16bf qa1 = load_a_frag(qb + (size_t)q0r * DHEAD, DHEAD, 32, lane);

    v8f o[4] = {};
    v8f mrow, lrow;
    #pragma unroll
    for (int r = 0; r < 8; ++r) { mrow[r] = -1e30f; lrow[r] = 0.0f; }

    int l = lane & 15, hi = lane >> 4;
    const float sc = 0.125f;  // 1/sqrt(64)

    for (int kblk = 0; kblk < SEQ; kblk += 64) {
        // ---- S = Q * K^T (16q x 64keys), two d-chunks of 32 ----
        v8f s[4] = {};
        #pragma unroll
        for (int kk = 0; kk < 2; ++kk) {
            v16bf aq = kk ? qa1 : qa0;
            #pragma unroll
            for (int j = 0; j < 4; ++j) {
                v16bf bk = load_b_frag(kb, DHEAD, kblk + j * 16, kk * 32, lane);
                s[j] = wmma_bf16(aq, bk, s[j]);
            }
        }
        // ---- online softmax (rows live across 16-lane halves) ----
        v8f mnew, scl;
        #pragma unroll
        for (int r = 0; r < 8; ++r) {
            s[0][r] *= sc; s[1][r] *= sc; s[2][r] *= sc; s[3][r] *= sc;
            float t = fmaxf(fmaxf(s[0][r], s[1][r]), fmaxf(s[2][r], s[3][r]));
            #pragma unroll
            for (int off = 1; off < 16; off <<= 1) t = fmaxf(t, __shfl_xor(t, off, 16));
            mnew[r] = fmaxf(mrow[r], t);
            scl[r]  = __expf(mrow[r] - mnew[r]);
        }
        #pragma unroll
        for (int r = 0; r < 8; ++r) {
            float p0 = __expf(s[0][r] - mnew[r]);
            float p1 = __expf(s[1][r] - mnew[r]);
            float p2 = __expf(s[2][r] - mnew[r]);
            float p3 = __expf(s[3][r] - mnew[r]);
            s[0][r] = p0; s[1][r] = p1; s[2][r] = p2; s[3][r] = p3;
            float t = (p0 + p1) + (p2 + p3);
            #pragma unroll
            for (int off = 1; off < 16; off <<= 1) t += __shfl_xor(t, off, 16);
            lrow[r] = lrow[r] * scl[r] + t;
            mrow[r] = mnew[r];
        }
        #pragma unroll
        for (int j = 0; j < 4; ++j)
            #pragma unroll
            for (int r = 0; r < 8; ++r) o[j][r] *= scl[r];

        // ---- restage P (16x64) through wave-private LDS into A-frags ----
        WAVE_MEM_FENCE();               // compiler fence; HW DS is in-order
        #pragma unroll
        for (int j = 0; j < 4; ++j)
            #pragma unroll
            for (int r = 0; r < 8; ++r)
                pl[(r + 8 * hi) * 64 + j * 16 + l] = (bf16)s[j][r];
        WAVE_MEM_FENCE();
        v16bf pa0 = load_a_frag(pl, 64, 0,  lane);
        v16bf pa1 = load_a_frag(pl, 64, 32, lane);

        // ---- O += P * V (two key-chunks of 32) ----
        #pragma unroll
        for (int kk = 0; kk < 2; ++kk) {
            v16bf pa = kk ? pa1 : pa0;
            #pragma unroll
            for (int j = 0; j < 4; ++j) {
                v16bf bv = load_b_frag(vb, SEQ, j * 16, kblk + kk * 32, lane);
                o[j] = wmma_bf16(pa, bv, o[j]);
            }
        }
    }

    // ---- epilogue: O * gate / l, write [b][n][h*64+d] bf16 ----
    #pragma unroll
    for (int r = 0; r < 8; ++r) {
        int row = q0r + r + 8 * hi;
        float g   = gates[((size_t)b * SEQ + row) * HEADS + h];
        float inv = g / fmaxf(lrow[r], 1e-20f);
        #pragma unroll
        for (int j = 0; j < 4; ++j)
            out[((size_t)b * SEQ + row) * (HEADS * DHEAD) + h * DHEAD + j * 16 + l]
                = (bf16)(o[j][r] * inv);
    }
}

// ---------------------------------------------------------------------------
// Host launch
// ---------------------------------------------------------------------------
extern "C" void kernel_launch(void* const* d_in, const int* in_sizes, int n_in,
                              void* d_out, int out_size, void* d_ws, size_t ws_size,
                              hipStream_t stream) {
    const float* x     = (const float*)d_in[0];
    const float* gamma = (const float*)d_in[1];
    const float* w_qkv = (const float*)d_in[2];
    const float* w_g   = (const float*)d_in[3];
    const float* b_g   = (const float*)d_in[4];
    const float* w_o   = (const float*)d_in[5];

    char* ws = (char*)d_ws;
    size_t off = 0;
    auto take = [&](size_t bytes) { void* p = ws + off; off = (off + bytes + 255) & ~(size_t)255; return p; };

    bf16*  xn    = (bf16*) take((size_t)ROWS * DIMX * 2);          // 8 MB
    float* gates = (float*)take((size_t)ROWS * HEADS * 4);         // 256 KB
    bf16*  wqkvT = (bf16*) take((size_t)NQKV * DIMX * 2);          // 1.5 MB
    bf16*  woT   = (bf16*) take((size_t)DIMX * DIMX * 2);          // 0.5 MB
    bf16*  qkv   = (bf16*) take((size_t)ROWS * NQKV * 2);          // 24 MB
    bf16*  qd    = (bf16*) take((size_t)ROWS * DHEAD * HEADS * 2); // 8 MB
    bf16*  kd    = (bf16*) take((size_t)ROWS * DHEAD * HEADS * 2); // 8 MB
    bf16*  vT    = (bf16*) take((size_t)ROWS * DHEAD * HEADS * 2); // 8 MB
    bf16*  attn  = (bf16*) take((size_t)ROWS * DIMX * 2);          // 8 MB
    (void)ws_size; (void)in_sizes; (void)n_in; (void)out_size;

    // 1. normalize + gates
    norm_gates_kernel<<<ROWS, 128, 0, stream>>>(x, gamma, w_g, b_g, xn, gates);

    // 2/3. weight transpose + bf16 convert
    transpose_w_kernel<<<(DIMX * NQKV + 255) / 256, 256, 0, stream>>>(w_qkv, wqkvT, DIMX, NQKV);
    transpose_w_kernel<<<(DIMX * DIMX + 255) / 256, 256, 0, stream>>>(w_o, woT, DIMX, DIMX);

    // 4. QKV GEMM: 8192 x 1536 x 512
    gemm_bf16_kernel<false><<<dim3(ROWS / 128, NQKV / 128), 256, 0, stream>>>(
        xn, wqkvT, qkv, ROWS, NQKV, DIMX);

    // 5. RoPE + split into attention layouts
    rope_split_kernel<<<(BATCH * HEADS * SEQ * 32) / 256, 256, 0, stream>>>(qkv, qd, kd, vT);

    // 6. flash attention + gate fusion
    attn_kernel<<<dim3(SEQ / 64, BATCH * HEADS), 128, 0, stream>>>(qd, kd, vT, gates, attn);

    // 7. output GEMM: 8192 x 512 x 512 -> fp32 d_out
    gemm_bf16_kernel<true><<<dim3(ROWS / 128, DIMX / 128), 256, 0, stream>>>(
        attn, woT, d_out, ROWS, DIMX, DIMX);
}
